// CausalAttention_39865886441792
// MI455X (gfx1250) — matrix operand order repaired and checked by reading, not compiled, not run on hardware
//
#include <hip/hip_runtime.h>

#define BB     4
#define SEQ    1024
#define MEML   1024
#define CTXL   2048
#define DMODEL 1024
#define NHEAD  16
#define DHEAD  64

typedef __attribute__((ext_vector_type(16))) __bf16 v16bf;
typedef __attribute__((ext_vector_type(8)))  __bf16 v8bf;
typedef __attribute__((ext_vector_type(8)))  float  v8f;
typedef __attribute__((ext_vector_type(4)))  unsigned int v4u;
typedef __attribute__((ext_vector_type(4)))  int v4i;
typedef __attribute__((ext_vector_type(8)))  int v8i;

#if defined(__has_builtin)
#if __has_builtin(__builtin_amdgcn_tensor_load_to_lds)
#define HAVE_TDM 1
#endif
#if __has_builtin(__builtin_amdgcn_s_wait_tensorcnt)
#define WAIT_TENSORCNT(n) __builtin_amdgcn_s_wait_tensorcnt(n)
#endif
#endif
#ifndef WAIT_TENSORCNT
#define WAIT_TENSORCNT(n) asm volatile("s_wait_tensorcnt %0" ::"i"(n) : "memory")
#endif

static __device__ __forceinline__ v8f wmma_bf16(v16bf a, v16bf b, v8f c) {
  // D = A(16x32 bf16) * B(32x16 bf16) + C(16x16 f32)
  return __builtin_amdgcn_wmma_f32_16x16x32_bf16(false, a, false, b, (short)0, c, false, false);
}

static __device__ __forceinline__ v8f zero_v8f() {
  v8f r;
#pragma unroll
  for (int i = 0; i < 8; ++i) r[i] = 0.0f;
  return r;
}
static __device__ __forceinline__ v16bf zero_v16bf() {
  v16bf r;
#pragma unroll
  for (int i = 0; i < 16; ++i) r[i] = (__bf16)0.0f;
  return r;
}
static __device__ __forceinline__ v16bf ld_frag(const __bf16* lo, const __bf16* hi) {
  v8bf a = *(const v8bf*)lo;
  v8bf b = *(const v8bf*)hi;
  v16bf r;
#pragma unroll
  for (int e = 0; e < 8; ++e) { r[e] = a[e]; r[8 + e] = b[e]; }
  return r;
}
// A-fragment (16x32): per-lane K = half*16 .. +15 contiguous
static __device__ __forceinline__ v16bf ld_a_frag(const __bf16* base) { return ld_frag(base, base + 8); }
// B-fragment (32x16): per-lane K = half*16 .. +15 contiguous
static __device__ __forceinline__ v16bf ld_b_frag(const __bf16* base) { return ld_frag(base, base + 8); }

// Generic LDS pointer -> LDS byte offset (aperture: offset is the low 32 address bits)
static __device__ __forceinline__ unsigned lds_off(const void* p) {
  return (unsigned)(unsigned long long)(size_t)p;
}

#ifdef HAVE_TDM
// Issue one 2D bf16 TDM tile load: tile (tile_d1 rows x tile_d0 elems), row pitch
// stride0 (elements), from gaddr into contiguous LDS at lds_addr. D# per ISA 8.3/8.4.
static __device__ __forceinline__ void tdm_load_2d(unsigned lds_addr, const void* gaddr,
                                                   unsigned tile_d0, unsigned tile_d1,
                                                   unsigned long long stride0_elems) {
  unsigned long long ga = (unsigned long long)(size_t)gaddr;
  v4u g0;
  g0[0] = 1u;                                                // count=1, user desc
  g0[1] = lds_addr;                                          // lds_addr (bytes)
  g0[2] = (unsigned)(ga & 0xffffffffu);                      // global_addr lo
  g0[3] = (unsigned)((ga >> 32) & 0x01ffffffu) | (2u << 30); // global_addr hi | type=2
  v8i g1;
  g1[0] = (int)(1u << 16);                                   // mask=0, data_size=2B
  g1[1] = (int)((tile_d0 & 0xffffu) << 16);                  // tensor_dim0 = tile_d0
  g1[2] = (int)(((tile_d0 >> 16) & 0xffffu) | ((tile_d1 & 0xffffu) << 16)); // tensor_dim1
  g1[3] = (int)(((tile_d1 >> 16) & 0xffffu) | ((tile_d0 & 0xffffu) << 16)); // tile_dim0
  g1[4] = (int)(tile_d1 & 0xffffu);                          // tile_dim1 (tile_dim2=0)
  g1[5] = (int)(unsigned)(stride0_elems & 0xffffffffu);      // tensor_dim0_stride lo
  g1[6] = (int)(unsigned)((stride0_elems >> 32) & 0xffffu);  // stride hi (dim1_stride=0)
  g1[7] = 0;
  v4i z4;
#pragma unroll
  for (int i = 0; i < 4; ++i) z4[i] = 0;
#if __clang_major__ >= 23
  v8i z8;
#pragma unroll
  for (int i = 0; i < 8; ++i) z8[i] = 0;
  __builtin_amdgcn_tensor_load_to_lds(g0, g1, z4, z4, z8, 0);
#else
  __builtin_amdgcn_tensor_load_to_lds(g0, g1, z4, z4, 0);
#endif
}
#endif

// ---------------- conversion kernels ----------------

__global__ void build_ctx_bf16(const float* __restrict__ mem, const float* __restrict__ x,
                               __bf16* __restrict__ ctx) {
  size_t total = (size_t)BB * CTXL * DMODEL;
  for (size_t idx = (size_t)blockIdx.x * blockDim.x + threadIdx.x; idx < total;
       idx += (size_t)gridDim.x * blockDim.x) {
    size_t d = idx % DMODEL;
    size_t r = (idx / DMODEL) % CTXL;
    size_t b = idx / ((size_t)DMODEL * CTXL);
    float v = (r < MEML) ? mem[(b * MEML + r) * DMODEL + d]
                         : x[(b * SEQ + (r - MEML)) * DMODEL + d];
    ctx[idx] = (__bf16)v;
  }
}

__global__ void cvt_bf16(const float* __restrict__ src, __bf16* __restrict__ dst, size_t n) {
  for (size_t idx = (size_t)blockIdx.x * blockDim.x + threadIdx.x; idx < n;
       idx += (size_t)gridDim.x * blockDim.x)
    dst[idx] = (__bf16)src[idx];
}

__global__ void cvt_transpose_bf16(const float* __restrict__ src, __bf16* __restrict__ dst,
                                   int K, int N) {
  size_t total = (size_t)K * N;
  for (size_t idx = (size_t)blockIdx.x * blockDim.x + threadIdx.x; idx < total;
       idx += (size_t)gridDim.x * blockDim.x) {
    size_t n = idx / K;
    size_t k = idx % K;
    dst[idx] = (__bf16)src[k * N + n];
  }
}

// ---------------- bf16 WMMA GEMM with TDM-staged, double-buffered LDS tiles -------------
// C(M x N) = A(M x K) * BT^T + bias ; BT is (N x K). Block = 256 thr = 8 waves,
// block tile 128 rows x 64 cols; per K-step TDM DMAs A(128x32) and B(64x32) tiles.
__global__ __launch_bounds__(256) void gemm_bf16_wmma(
    const __bf16* __restrict__ A, const __bf16* __restrict__ BT,
    const float* __restrict__ bias, float* __restrict__ Cf, __bf16* __restrict__ Cbf,
    int M, int N, int K, int rowDiv, int rowPad, int rowOff) {
  __shared__ __attribute__((aligned(16))) __bf16 atile[2][128][32];
  __shared__ __attribute__((aligned(16))) __bf16 btile[2][64][32];

  int w = threadIdx.x >> 5;
  int lane = threadIdx.x & 31;
  int half = lane >> 4;
  int l16 = lane & 15;
  int m0 = blockIdx.y * 128;
  int n0 = blockIdx.x * 64;
  int mw = w * 16;

  // block-uniform A row remap (rows m0..m0+127 stay in one rowDiv chunk)
  const __bf16* Abase = A + (size_t)(m0 + (m0 / rowDiv) * rowPad + rowOff) * K;
  const __bf16* Bbase = BT + (size_t)n0 * K;
  const int nk = K / 32;

#ifdef HAVE_TDM
  if (w == 0) {
    tdm_load_2d(lds_off(&atile[0][0][0]), Abase, 32, 128, (unsigned long long)K);
    tdm_load_2d(lds_off(&btile[0][0][0]), Bbase, 32, 64, (unsigned long long)K);
  }
#endif

  v8f acc[4];
#pragma unroll
  for (int t = 0; t < 4; ++t) acc[t] = zero_v8f();

  for (int ks = 0; ks < nk; ++ks) {
    int buf = ks & 1;
    __syncthreads();
#ifdef HAVE_TDM
    if (w == 0) {
      if (ks + 1 < nk) {
        int k0n = (ks + 1) * 32;
        tdm_load_2d(lds_off(&atile[buf ^ 1][0][0]), Abase + k0n, 32, 128, (unsigned long long)K);
        tdm_load_2d(lds_off(&btile[buf ^ 1][0][0]), Bbase + k0n, 32, 64, (unsigned long long)K);
        WAIT_TENSORCNT(2);   // previous tile complete (TDM is in-order per wave)
      } else {
        WAIT_TENSORCNT(0);
      }
    }
#else
    {
      int k0 = ks * 32;
      for (int u = threadIdx.x; u < 512; u += 256) {
        int row = u >> 2, seg = u & 3;
        *(v8bf*)&atile[buf][row][seg * 8] = *(const v8bf*)(Abase + (size_t)row * K + k0 + seg * 8);
      }
      for (int u = threadIdx.x; u < 256; u += 256) {
        int row = u >> 2, seg = u & 3;
        *(v8bf*)&btile[buf][row][seg * 8] = *(const v8bf*)(Bbase + (size_t)row * K + k0 + seg * 8);
      }
    }
#endif
    __syncthreads();

    // batch all fragment loads, then issue WMMAs back-to-back
    v16bf a = ld_frag(&atile[buf][mw + l16][half * 16], &atile[buf][mw + l16][half * 16 + 8]);
    v16bf bfr[4];
#pragma unroll
    for (int t = 0; t < 4; ++t)
      bfr[t] = ld_frag(&btile[buf][t * 16 + l16][half * 16], &btile[buf][t * 16 + l16][half * 16 + 8]);
#pragma unroll
    for (int t = 0; t < 4; ++t) acc[t] = wmma_bf16(a, bfr[t], acc[t]);
  }

#pragma unroll
  for (int t = 0; t < 4; ++t) {
    int col = n0 + t * 16 + l16;
    float bv = bias ? bias[col] : 0.0f;
#pragma unroll
    for (int e = 0; e < 8; ++e) {
      int row = m0 + mw + e + half * 8;
      float val = acc[t][e] + bv;
      size_t idx = (size_t)row * N + col;
      if (Cf)  Cf[idx]  = val;
      if (Cbf) Cbf[idx] = (__bf16)val;
    }
  }
}

// ---------------- flash attention with rel-shift (Transformer-XL) ----------------
// Grid (SEQ/64, NHEAD, BB), 128 thr = 4 waves; wave w owns query rows i0+16w..+15.
// K/V 32-key tiles staged by TDM (double-buffered); all matmuls on WMMA;
// rel-shift handled as a WMMA Toeplitz band + LDS diagonal select.
__global__ __launch_bounds__(128) void flash_attn_relshift(
    const __bf16* __restrict__ Q, const __bf16* __restrict__ KV,
    const __bf16* __restrict__ PR, __bf16* __restrict__ O) {
  const float scale = 0.125f;  // DHEAD^-0.5
  int b = blockIdx.z, h = blockIdx.y;
  int i0 = blockIdx.x * 64;
  int w = threadIdx.x >> 5;
  int lane = threadIdx.x & 31;
  int half = lane >> 4;
  int l16 = lane & 15;
  int i0w = i0 + w * 16;

  __shared__ __attribute__((aligned(16))) __bf16 kraw[2][32][64];  // K tile [key][dh]
  __shared__ __attribute__((aligned(16))) __bf16 vraw[2][32][64];  // V tile [key][dh]
  __shared__ __attribute__((aligned(16))) __bf16 vt[64][48];       // V transposed [dh][key]
  __shared__ __attribute__((aligned(16))) __bf16 pstage[4][16][32];
  __shared__ __attribute__((aligned(16))) float  spband[4][16][48];

  const __bf16* qp = Q + ((size_t)(b * SEQ + i0w + l16)) * DMODEL + h * DHEAD;
  v16bf qa[2];
  qa[0] = ld_a_frag(qp + 0 + half * 16);
  qa[1] = ld_a_frag(qp + 32 + half * 16);

  v8f oacc[4];
#pragma unroll
  for (int t = 0; t < 4; ++t) oacc[t] = zero_v8f();
  float mrow[8], lrow[8];
#pragma unroll
  for (int e = 0; e < 8; ++e) { mrow[e] = -3.0e38f; lrow[e] = 0.0f; }

  const int jEnd = MEML + i0 + 64;
  const int nT = jEnd / 32;
  const __bf16* kvrow0 = KV + ((size_t)(b * CTXL)) * (2 * DMODEL) + h * DHEAD;

#ifdef HAVE_TDM
  if (w == 0) {
    tdm_load_2d(lds_off(&kraw[0][0][0]), kvrow0, 64, 32, 2 * DMODEL);
    tdm_load_2d(lds_off(&vraw[0][0][0]), kvrow0 + DMODEL, 64, 32, 2 * DMODEL);
  }
#endif

  for (int jt = 0; jt < nT; ++jt) {
    int jj0 = jt * 32;
    int buf = jt & 1;
    __syncthreads();
#ifdef HAVE_TDM
    if (w == 0) {
      if (jt + 1 < nT) {
        const __bf16* nb = kvrow0 + (size_t)(jj0 + 32) * (2 * DMODEL);
        tdm_load_2d(lds_off(&kraw[buf ^ 1][0][0]), nb, 64, 32, 2 * DMODEL);
        tdm_load_2d(lds_off(&vraw[buf ^ 1][0][0]), nb + DMODEL, 64, 32, 2 * DMODEL);
        WAIT_TENSORCNT(2);
      } else {
        WAIT_TENSORCNT(0);
      }
    }
#else
    for (int u = threadIdx.x; u < 256; u += 128) {
      int key = u >> 3, ch = u & 7;
      const __bf16* src = kvrow0 + (size_t)(jj0 + key) * (2 * DMODEL);
      *(v8bf*)&kraw[buf][key][ch * 8] = *(const v8bf*)(src + ch * 8);
      *(v8bf*)&vraw[buf][key][ch * 8] = *(const v8bf*)(src + DMODEL + ch * 8);
    }
#endif
    __syncthreads();

    // transpose V tile LDS->LDS for the P@V B-fragments
    for (int u = threadIdx.x; u < 256; u += 128) {
      int key = u >> 3, ch = u & 7;
      v8bf vv = *(const v8bf*)&vraw[buf][key][ch * 8];
#pragma unroll
      for (int e = 0; e < 8; ++e) vt[ch * 8 + e][key] = vv[e];
    }
    __syncthreads();

    // ---- S = scale * Q K^T  (batch fragments, then WMMAs back-to-back) ----
    v16bf kb[4];
#pragma unroll
    for (int ct = 0; ct < 2; ++ct)
#pragma unroll
      for (int ks = 0; ks < 2; ++ks)
        kb[ct * 2 + ks] = ld_b_frag(&kraw[buf][ct * 16 + l16][ks * 32 + half * 16]);
    v8f s[2];
#pragma unroll
    for (int ct = 0; ct < 2; ++ct) {
      v8f a = zero_v8f();
      a = wmma_bf16(qa[0], kb[ct * 2 + 0], a);
      a = wmma_bf16(qa[1], kb[ct * 2 + 1], a);
#pragma unroll
      for (int e = 0; e < 8; ++e) s[ct][e] = a[e] * scale;
    }

    // ---- positional Toeplitz band via WMMA ----
    bool posActive = (jj0 + 31 >= MEML);
    if (posActive) {
      int rbase = SEQ - 16 - i0w + (jj0 - MEML);
#pragma unroll
      for (int ct = 0; ct < 3; ++ct) {
        int r = rbase + ct * 16 + l16;
        bool rok = (r >= 0) && (r < SEQ);
        v16bf pb0 = rok ? ld_b_frag(PR + (size_t)r * DHEAD + half * 16) : zero_v16bf();
        v16bf pb1 = rok ? ld_b_frag(PR + (size_t)r * DHEAD + 32 + half * 16) : zero_v16bf();
        v8f sp = zero_v8f();
        sp = wmma_bf16(qa[0], pb0, sp);
        sp = wmma_bf16(qa[1], pb1, sp);
#pragma unroll
        for (int e = 0; e < 8; ++e) spband[w][e + half * 8][ct * 16 + l16] = sp[e] * scale;
      }
      asm volatile("s_wait_dscnt 0" ::: "memory");
    }

    // ---- add shifted pos term + causal mask ----
#pragma unroll
    for (int ct = 0; ct < 2; ++ct) {
#pragma unroll
      for (int e = 0; e < 8; ++e) {
        int ii = e + half * 8;
        int i = i0w + ii;
        int jc = ct * 16 + l16;
        int jj = jj0 + jc;
        float v = s[ct][e];
        if (jj > i + MEML)   v = -3.0e38f;
        else if (jj >= MEML) v += spband[w][ii][15 - ii + jc];
        s[ct][e] = v;
      }
    }

    // ---- online softmax ----
    float sc8[8];
#pragma unroll
    for (int e = 0; e < 8; ++e) {
      float t = fmaxf(s[0][e], s[1][e]);
#pragma unroll
      for (int off = 1; off < 16; off <<= 1) t = fmaxf(t, __shfl_xor(t, off, 32));
      float mnew = fmaxf(mrow[e], t);
      float sc = __expf(mrow[e] - mnew);
      float p0 = __expf(s[0][e] - mnew);
      float p1 = __expf(s[1][e] - mnew);
      float rs = p0 + p1;
#pragma unroll
      for (int off = 1; off < 16; off <<= 1) rs += __shfl_xor(rs, off, 32);
      lrow[e] = lrow[e] * sc + rs;
      mrow[e] = mnew;
      sc8[e] = sc;
      int ii = e + half * 8;
      pstage[w][ii][l16]      = (__bf16)p0;
      pstage[w][ii][16 + l16] = (__bf16)p1;
    }
    asm volatile("s_wait_dscnt 0" ::: "memory");

    // ---- O = O*sc + P V  (batch fragments, then WMMAs) ----
    v16bf pa = ld_frag(&pstage[w][l16][half * 16], &pstage[w][l16][half * 16 + 8]);
    v16bf vb[4];
#pragma unroll
    for (int t = 0; t < 4; ++t) {
      int dh = t * 16 + l16;
      vb[t] = ld_frag(&vt[dh][half * 16], &vt[dh][half * 16 + 8]);
    }
#pragma unroll
    for (int t = 0; t < 4; ++t) {
      v8f o = oacc[t];
#pragma unroll
      for (int e = 0; e < 8; ++e) o[e] *= sc8[e];
      oacc[t] = wmma_bf16(pa, vb[t], o);
    }
  }

#pragma unroll
  for (int t = 0; t < 4; ++t) {
#pragma unroll
    for (int e = 0; e < 8; ++e) {
      int ii = e + half * 8;
      float val = oacc[t][e] / lrow[e];
      O[((size_t)(b * SEQ + i0w + ii)) * DMODEL + h * DHEAD + t * 16 + l16] = (__bf16)val;
    }
  }
}

// ---------------- host launch ----------------

extern "C" void kernel_launch(void* const* d_in, const int* in_sizes, int n_in,
                              void* d_out, int out_size, void* d_ws, size_t ws_size,
                              hipStream_t stream) {
  const float* x    = (const float*)d_in[0];
  const float* pose = (const float*)d_in[1];
  const float* mem  = (const float*)d_in[2];
  const float* Wq   = (const float*)d_in[3];
  const float* bq   = (const float*)d_in[4];
  const float* Wkv  = (const float*)d_in[5];
  const float* bkv  = (const float*)d_in[6];
  const float* Wpos = (const float*)d_in[7];
  const float* bpos = (const float*)d_in[8];
  const float* Wout = (const float*)d_in[9];
  const float* bout = (const float*)d_in[10];
  (void)in_sizes; (void)n_in; (void)out_size; (void)ws_size;

  char* ws = (char*)d_ws;
  size_t off = 0;
  auto take = [&](size_t elems) {
    char* p = ws + off;
    off = (off + elems * sizeof(__bf16) + 255) & ~(size_t)255;
    return (__bf16*)p;
  };
  __bf16* ctx_bf  = take((size_t)BB * CTXL * DMODEL);
  __bf16* wqT     = take((size_t)DMODEL * DMODEL);
  __bf16* wkvT    = take((size_t)2 * DMODEL * DMODEL);
  __bf16* wposT   = take((size_t)DHEAD * DMODEL);
  __bf16* woutT   = take((size_t)DMODEL * DMODEL);
  __bf16* pose_bf = take((size_t)SEQ * DMODEL);
  __bf16* q_bf    = take((size_t)BB * SEQ * DMODEL);
  __bf16* kv_bf   = take((size_t)BB * CTXL * 2 * DMODEL);
  __bf16* posr_bf = take((size_t)SEQ * DHEAD);
  __bf16* o_bf    = take((size_t)BB * SEQ * DMODEL);

  const int IDENT = 1 << 30;

  build_ctx_bf16<<<4096, 256, 0, stream>>>(mem, x, ctx_bf);
  cvt_bf16<<<2048, 256, 0, stream>>>(pose, pose_bf, (size_t)SEQ * DMODEL);
  cvt_transpose_bf16<<<2048, 256, 0, stream>>>(Wq,   wqT,   DMODEL, DMODEL);
  cvt_transpose_bf16<<<2048, 256, 0, stream>>>(Wkv,  wkvT,  DMODEL, 2 * DMODEL);
  cvt_transpose_bf16<<<512,  256, 0, stream>>>(Wpos, wposT, DMODEL, DHEAD);
  cvt_transpose_bf16<<<2048, 256, 0, stream>>>(Wout, woutT, DMODEL, DMODEL);

  // q = x@Wq + bq  (A rows are the x-half of ctx)
  gemm_bf16_wmma<<<dim3(DMODEL / 64, (BB * SEQ) / 128), 256, 0, stream>>>(
      ctx_bf, wqT, bq, nullptr, q_bf, BB * SEQ, DMODEL, DMODEL, 1024, 1024, 1024);
  // kv = ctx@Wkv + bkv
  gemm_bf16_wmma<<<dim3((2 * DMODEL) / 64, (BB * CTXL) / 128), 256, 0, stream>>>(
      ctx_bf, wkvT, bkv, nullptr, kv_bf, BB * CTXL, 2 * DMODEL, DMODEL, IDENT, 0, 0);
  // pos = pos_emb@Wpos + bpos
  gemm_bf16_wmma<<<dim3(DHEAD / 64, SEQ / 128), 256, 0, stream>>>(
      pose_bf, wposT, bpos, nullptr, posr_bf, SEQ, DHEAD, DMODEL, IDENT, 0, 0);

  flash_attn_relshift<<<dim3(SEQ / 64, NHEAD, BB), 128, 0, stream>>>(q_bf, kv_bf, posr_bf, o_bf);

  // out = O@Wout + bout -> fp32
  gemm_bf16_wmma<<<dim3(DMODEL / 64, (BB * SEQ) / 128), 256, 0, stream>>>(
      o_bf, woutT, bout, (float*)d_out, nullptr, BB * SEQ, DMODEL, DMODEL, IDENT, 0, 0);
}
